// RelationalSemiringComposer_4243427688874
// MI455X (gfx1250) — matrix operand order repaired and checked by reading, not compiled, not run on hardware
//
#include <hip/hip_runtime.h>
#include <hip/hip_bf16.h>

typedef __bf16 bf16;
typedef __attribute__((ext_vector_type(16))) __bf16 v16bf;
typedef __attribute__((ext_vector_type(8)))  __bf16 v8bf;
typedef __attribute__((ext_vector_type(4)))  __bf16 v4bf;
typedef __attribute__((ext_vector_type(8)))  float  v8f;

#define MREL 8
#define DM   256
#define ROWS 32
#define NTHREADS 256
#define SMEM_BYTES 184576

__device__ __forceinline__ v8f wmma_bf16(v16bf a, v16bf b, v8f c) {
  return __builtin_amdgcn_wmma_f32_16x16x32_bf16(false, a, false, b, (short)0, c,
                                                 false, false);
}

// A operand (16x32 bf16): lane L holds row M=L%16; elements 0..15 = K(kbase+16*half+e).
// Two contiguous 16B LDS loads per fragment.
__device__ __forceinline__ v16bf load_a_frag(const bf16* __restrict__ row, int kbase, int half) {
  const bf16* p0 = row + kbase + (half << 4);
  v8bf lo = *(const v8bf*)(p0);
  v8bf hi = *(const v8bf*)(p0 + 8);
  v16bf a;
#pragma unroll
  for (int e = 0; e < 8; ++e) { a[e] = lo[e]; a[e + 8] = hi[e]; }
  return a;
}

// B operand (32x16 bf16): lane L holds col N=L%16, K = kbase + 16*half + e.
// Weights are pre-transposed to [N][K] so this is one contiguous 32B global load.
__device__ __forceinline__ v16bf load_b_frag(const bf16* __restrict__ colrow, int kbase, int half) {
  return *(const v16bf*)(colrow + kbase + (half << 4));
}

// ---------------- weight prep: fp32 -> bf16, transpose to [out][in] ----------------
__global__ void prep_weights(const float* __restrict__ Wmax,
                             const float* __restrict__ Wphi,
                             const float* __restrict__ Wc,
                             bf16* __restrict__ WmaxT,
                             bf16* __restrict__ WphiT,
                             bf16* __restrict__ WcT) {
  const int total_mm = MREL * DM * DM;   // 524288
  const int total_wc = DM * 3 * DM;      // 196608
  for (int idx = blockIdx.x * blockDim.x + threadIdx.x;
       idx < total_mm + total_wc;
       idx += gridDim.x * blockDim.x) {
    if (idx < total_mm) {
      int m = idx >> 16;
      int r = idx & 65535;
      int n = r >> 8;            // out col
      int k = r & 255;           // in row
      int src = (m << 16) + (k << 8) + n;   // W[m][k][n]
      WmaxT[idx] = (bf16)Wmax[src];
      WphiT[idx] = (bf16)Wphi[src];
    } else {
      int i = idx - total_mm;
      int n = i / 768;
      int k = i - n * 768;
      WcT[i] = (bf16)Wc[k * DM + n];     // Wc[k][n] -> WcT[n][k]
    }
  }
}

// ---------------- fused composer ----------------
__global__ __launch_bounds__(NTHREADS, 1)
void fused_composer(const float* __restrict__ zs,
                    const float* __restrict__ Wb,
                    const float* __restrict__ bb,
                    const float* __restrict__ bmax,
                    const float* __restrict__ bphi,
                    const float* __restrict__ bc,
                    const float* __restrict__ gamma,
                    const float* __restrict__ beta,
                    const bf16* __restrict__ WmaxT,
                    const bf16* __restrict__ WphiT,
                    const bf16* __restrict__ WcT,
                    float* __restrict__ out,
                    int N)
{
  extern __shared__ char smem[];
  bf16*  s_zs    = (bf16*)smem;                  // [8][32][256] bf16  (131072 B)
  bf16*  s_cat   = (bf16*)(smem + 131072);       // [32][768]  bf16    (49152 B)
  float* s_logit = (float*)(smem + 180224);      // [32][8]
  float* s_gate  = (float*)(smem + 181248);      // [32][8]
  float* s_red   = (float*)(smem + 182272);      // [32][8][2]
  float* s_mu    = (float*)(smem + 184320);      // [32]
  float* s_rstd  = (float*)(smem + 184448);      // [32]
  float* zb      = (float*)smem;                 // [32][256] f32, aliases s_zs (phase>=4)

  const int tid  = threadIdx.x;
  const int lane = tid & 31;
  const int wave = tid >> 5;
  const int half = lane >> 4;
  const int l15  = lane & 15;
  const int rowsub = wave >> 2;        // 0..1 : which 16-row subtile
  const int cgrp   = wave & 3;         // 0..3 : column-tile group
  const long row0 = (long)blockIdx.x * ROWS;

  // ---- Phase 0: stage zs tile to LDS as bf16 (float4-coalesced) ----
#pragma unroll 4
  for (int it = 0; it < 64; ++it) {
    int s = tid + it * NTHREADS;       // 16384 float4 slots
    int base = s << 2;
    int m = base >> 13;
    int r = (base >> 8) & (ROWS - 1);
    int c = base & (DM - 1);
    long n = row0 + r;
    float4 v = make_float4(0.f, 0.f, 0.f, 0.f);
    if (n < N) v = *(const float4*)(zs + (((long)m * N + n) << 8) + c);
    v4bf b; b[0] = (bf16)v.x; b[1] = (bf16)v.y; b[2] = (bf16)v.z; b[3] = (bf16)v.w;
    *(v4bf*)(s_zs + base) = b;
  }
  __syncthreads();

  // ---- Phase 1: gate logits (thread = (row=lane, j=wave)) ----
  {
    const int n_l = lane;
    const int j   = wave;
    float acc = bb[j];
#pragma unroll
    for (int m = 0; m < MREL; ++m) {
      const bf16* zr = s_zs + ((m * ROWS + n_l) << 8);
      const float* wb = Wb + ((m * DM) << 3) + j;
#pragma unroll 4
      for (int d = 0; d < DM; ++d) acc += (float)zr[d] * wb[d << 3];
    }
    s_logit[n_l * MREL + j] = acc;
  }
  __syncthreads();
  if (tid < ROWS) {
    float l[MREL];
    float mx = -3.402823e38f;
#pragma unroll
    for (int j = 0; j < MREL; ++j) { l[j] = s_logit[tid * MREL + j]; mx = fmaxf(mx, l[j]); }
    float s = 0.f;
#pragma unroll
    for (int j = 0; j < MREL; ++j) { l[j] = __expf(l[j] - mx); s += l[j]; }
    float inv = 1.f / s;
#pragma unroll
    for (int j = 0; j < MREL; ++j) s_gate[tid * MREL + j] = l[j] * inv;
  }
  __syncthreads();

  // ---- Phase 1.5: z_sum -> s_cat[:, 0:256] ----
#pragma unroll 4
  for (int it = 0; it < 32; ++it) {
    int idx = tid + it * NTHREADS;     // 0..8191
    int n_l = idx >> 8;
    int d   = idx & (DM - 1);
    float s = 0.f;
#pragma unroll
    for (int m = 0; m < MREL; ++m)
      s += s_gate[n_l * MREL + m] * (float)s_zs[((m * ROWS + n_l) << 8) + d];
    s_cat[n_l * 768 + d] = (bf16)s;
  }

  // ---- Phase 2: max & phi branches via WMMA bf16 (f32 accum) ----
  float vmax[4][8], s1[4][8], sq[4][8];
#pragma unroll
  for (int i = 0; i < 4; ++i)
#pragma unroll
    for (int v = 0; v < 8; ++v) { vmax[i][v] = -3.402823e38f; s1[i][v] = 0.f; sq[i][v] = 0.f; }

  const bf16* arow_base = s_zs + (((rowsub << 4) + l15) << 8);

#pragma unroll 1
  for (int m = 0; m < MREL; ++m) {
    const bf16* arow = arow_base + ((m * ROWS) << 8);
    v8f am[4], ap[4];
#pragma unroll
    for (int i = 0; i < 4; ++i) {            // fold biases into accumulators
      int col = ((cgrp + (i << 2)) << 4) + l15;
      float b1 = bmax[(m << 8) + col];
      float b2 = bphi[(m << 8) + col];
      v8f x, y;
#pragma unroll
      for (int v = 0; v < 8; ++v) { x[v] = b1; y[v] = b2; }
      am[i] = x; ap[i] = y;
    }
#pragma unroll 1
    for (int k = 0; k < 8; ++k) {
      v16bf A = load_a_frag(arow, k << 5, half);
#pragma unroll
      for (int i = 0; i < 4; ++i) {
        int col = ((cgrp + (i << 2)) << 4) + l15;
        v16bf Bm = load_b_frag(WmaxT + ((long)m << 16) + (col << 8), k << 5, half);
        am[i] = wmma_bf16(A, Bm, am[i]);
        v16bf Bp = load_b_frag(WphiT + ((long)m << 16) + (col << 8), k << 5, half);
        ap[i] = wmma_bf16(A, Bp, ap[i]);
      }
    }
#pragma unroll
    for (int i = 0; i < 4; ++i)
#pragma unroll
      for (int v = 0; v < 8; ++v) {
        vmax[i][v] = fmaxf(vmax[i][v], am[i][v]);
        float ph = ap[i][v];
        s1[i][v] += ph;
        sq[i][v] += ph * ph;
      }
  }

  // ---- Phase 3: z_max, z_agr = 0.5*((sum phi)^2 - sum phi^2) -> s_cat ----
#pragma unroll
  for (int i = 0; i < 4; ++i) {
    int col = ((cgrp + (i << 2)) << 4) + l15;
#pragma unroll
    for (int v = 0; v < 8; ++v) {
      int rl = (rowsub << 4) + v + (half << 3);   // C/D layout: M = v + 8*half
      s_cat[rl * 768 + 256 + col] = (bf16)vmax[i][v];
      s_cat[rl * 768 + 512 + col] = (bf16)(0.5f * (s1[i][v] * s1[i][v] - sq[i][v]));
    }
  }
  __syncthreads();

  // ---- Phase 4: combine GEMM  z = cat[32x768] @ Wc + bc ----
  {
    const bf16* crow = s_cat + ((rowsub << 4) + l15) * 768;
    v8f acc[4];
#pragma unroll
    for (int i = 0; i < 4; ++i) {
      int col = ((cgrp + (i << 2)) << 4) + l15;
      float b = bc[col];
      v8f x;
#pragma unroll
      for (int v = 0; v < 8; ++v) x[v] = b;
      acc[i] = x;
    }
#pragma unroll 1
    for (int k = 0; k < 24; ++k) {
      v16bf A = load_a_frag(crow, k << 5, half);
#pragma unroll
      for (int i = 0; i < 4; ++i) {
        int col = ((cgrp + (i << 2)) << 4) + l15;
        v16bf B = load_b_frag(WcT + col * 768, k << 5, half);
        acc[i] = wmma_bf16(A, B, acc[i]);
      }
    }
#pragma unroll
    for (int i = 0; i < 4; ++i) {
      int col = ((cgrp + (i << 2)) << 4) + l15;
#pragma unroll
      for (int v = 0; v < 8; ++v) {
        int rl = (rowsub << 4) + v + (half << 3);
        zb[(rl << 8) + col] = acc[i][v];
      }
    }
  }
  __syncthreads();

  // ---- Phase 5: LayerNorm over D=256 per row ----
  {
    int rl  = tid >> 3;
    int seg = tid & 7;
    float s = 0.f, s2 = 0.f;
#pragma unroll 4
    for (int c = seg * 32; c < seg * 32 + 32; ++c) {
      float v = zb[(rl << 8) + c];
      s += v; s2 += v * v;
    }
    s_red[(tid << 1) + 0] = s;
    s_red[(tid << 1) + 1] = s2;
  }
  __syncthreads();
  if (tid < ROWS) {
    float s = 0.f, s2 = 0.f;
#pragma unroll
    for (int g = 0; g < 8; ++g) {
      s  += s_red[((tid * 8 + g) << 1) + 0];
      s2 += s_red[((tid * 8 + g) << 1) + 1];
    }
    float mu  = s * (1.f / DM);
    float var = s2 * (1.f / DM) - mu * mu;
    s_mu[tid]   = mu;
    s_rstd[tid] = rsqrtf(var + 1e-5f);
  }
  __syncthreads();
#pragma unroll 4
  for (int it = 0; it < 32; ++it) {
    int idx = tid + it * NTHREADS;
    int rl  = idx >> 8;
    int c   = idx & (DM - 1);
    long n  = row0 + rl;
    if (n < N) {
      float v = (zb[idx] - s_mu[rl]) * s_rstd[rl] * gamma[c] + beta[c];
      out[(n << 8) + c] = v;
    }
  }
}

extern "C" void kernel_launch(void* const* d_in, const int* in_sizes, int n_in,
                              void* d_out, int out_size, void* d_ws, size_t ws_size,
                              hipStream_t stream) {
  (void)n_in; (void)out_size; (void)ws_size;
  const float* zs    = (const float*)d_in[0];
  const float* Wb    = (const float*)d_in[1];
  const float* bb    = (const float*)d_in[2];
  const float* Wmax  = (const float*)d_in[3];
  const float* bmax  = (const float*)d_in[4];
  const float* Wphi  = (const float*)d_in[5];
  const float* bphi  = (const float*)d_in[6];
  const float* Wc    = (const float*)d_in[7];
  const float* bc    = (const float*)d_in[8];
  const float* gamma = (const float*)d_in[9];
  const float* beta  = (const float*)d_in[10];
  float* out = (float*)d_out;

  int N = in_sizes[0] / (MREL * DM);

  // Workspace: bf16 transposed weights (2,490,368 bytes total)
  char* ws = (char*)d_ws;
  bf16* WmaxT = (bf16*)(ws);
  bf16* WphiT = (bf16*)(ws + 1048576);
  bf16* WcT   = (bf16*)(ws + 2097152);

  prep_weights<<<512, 256, 0, stream>>>(Wmax, Wphi, Wc, WmaxT, WphiT, WcT);

  int nblocks = (N + ROWS - 1) / ROWS;
  fused_composer<<<nblocks, NTHREADS, SMEM_BYTES, stream>>>(
      zs, Wb, bb, bmax, bphi, bc, gamma, beta, WmaxT, WphiT, WcT, out, N);
}
